// AspectSent_67035849556495
// MI455X (gfx1250) — matrix operand order repaired and checked by reading, not compiled, not run on hardware
//
#include <hip/hip_runtime.h>
#include <hip/hip_bf16.h>
#include <stdint.h>

// ---------------- problem constants (match reference) ----------------
constexpr int Bn   = 256;
constexpr int Tn   = 128;
constexpr int EMB  = 300;
constexpr int MDIM = 50;
constexpr int DTOT = 350;   // EMB + MDIM
constexpr int KP   = 352;   // DTOT padded to multiple of 32 for WMMA K
constexpr int Hn   = 1024;
constexpr int HDn  = 512;
constexpr int Gn   = 1536;  // 3*HD gates
constexpr float C1 = 1.0f, C2 = 1.0f;

// ---------------- CDNA5 feature detection ----------------
#if defined(__has_builtin)
#if __has_builtin(__builtin_amdgcn_tensor_load_to_lds) && \
    __has_builtin(__builtin_amdgcn_s_wait_tensorcnt)
#define HAVE_TDM 1
#endif
#endif
#ifndef HAVE_TDM
#define HAVE_TDM 0
#endif

// ---------------- WMMA types ----------------
typedef __attribute__((ext_vector_type(16))) __bf16 v16bf;
typedef __attribute__((ext_vector_type(8)))  float  v8f;
#if HAVE_TDM
typedef unsigned int u32x4 __attribute__((ext_vector_type(4)));
typedef int          i32x4 __attribute__((ext_vector_type(4)));
typedef int          i32x8 __attribute__((ext_vector_type(8)));
#endif

union FragBF { uint4 u[2]; v16bf v; };
static_assert(sizeof(FragBF) == 32, "frag size");

__device__ inline v8f wmma_bf16(v16bf a, v16bf b, v8f c) {
  // D = A(16x32 bf16) * B(32x16 bf16) + C(16x16 f32)
  return __builtin_amdgcn_wmma_f32_16x16x32_bf16(
      false, a, false, b, (short)0, c, false, false);
}

__device__ inline unsigned short f2bf(float f) {
  unsigned int u = __float_as_uint(f);
  unsigned int r = (u + 0x7FFFu + ((u >> 16) & 1u)) >> 16;
  return (unsigned short)r;
}
__device__ inline float bf2f(unsigned short u) {
  return __uint_as_float(((unsigned)u) << 16);
}

#if HAVE_TDM
// Issue one 2D Tensor-Data-Mover load: tile_d0 x tile_d1 elements (2 bytes
// each), tensor line stride stride0 (elements), into contiguous LDS at `lds`.
__device__ inline void tdm_load_2d(const void* gbase, size_t byte_off, void* lds,
                                   unsigned tile_d0, unsigned tile_d1,
                                   unsigned tensor_d0, unsigned tensor_d1,
                                   unsigned stride0) {
  unsigned long long ga = (unsigned long long)(uintptr_t)gbase + byte_off;
  u32x4 g0 = {1u,                                   // count=1
              (unsigned)(uintptr_t)lds,             // lds_addr
              (unsigned)(ga & 0xFFFFFFFFu),
              (unsigned)((ga >> 32) & 0x1FFFFFFu) | (2u << 30)};  // type=2
  i32x8 g1 = {0x00010000,                            // data_size = 2 bytes
              (int)((tensor_d0 & 0xFFFFu) << 16),    // tensor_dim0 lo
              (int)((tensor_d0 >> 16) | ((tensor_d1 & 0xFFFFu) << 16)),
              (int)((tensor_d1 >> 16) | (tile_d0 << 16)),
              (int)(tile_d1 & 0xFFFFu),
              (int)stride0,
              0, 0};
  i32x4 z4 = {0, 0, 0, 0};
#if __clang_major__ >= 23
  i32x8 z8 = {0, 0, 0, 0, 0, 0, 0, 0};
  __builtin_amdgcn_tensor_load_to_lds(g0, g1, z4, z4, z8, 0);
#else
  __builtin_amdgcn_tensor_load_to_lds(g0, g1, z4, z4, 0);
#endif
}
#endif

// ---------------- utility kernels ----------------
__global__ void zero_f32_kernel(float* p, size_t n) {
  size_t i = (size_t)blockIdx.x * blockDim.x + threadIdx.x;
  size_t s = (size_t)gridDim.x * blockDim.x;
  for (; i < n; i += s) p[i] = 0.0f;
}

// Gather embeddings -> x_bf [B*T, KP] (bf16) and reversed-valid x -> xr_bf
__global__ void embed_kernel(const int* __restrict__ sents,
                             const int* __restrict__ masks,
                             const int* __restrict__ lens,
                             const float* __restrict__ wemb,
                             const float* __restrict__ memb,
                             unsigned short* __restrict__ xbf,
                             unsigned short* __restrict__ xrbf) {
  int bt = blockIdx.x;
  int b = bt / Tn, t = bt % Tn;
  int len = lens[b];
  bool valid = (t < len);
  int ts = len - 1 - t;
  int sid = sents[b * Tn + t];
  int mid = masks[b * Tn + t];
  int sid_r = 0, mid_r = 0;
  if (valid) { sid_r = sents[b * Tn + ts]; mid_r = masks[b * Tn + ts]; }
  size_t base = (size_t)bt * KP;
  for (int d = threadIdx.x; d < KP; d += blockDim.x) {
    float v = 0.f, vr = 0.f;
    if (d < EMB) {
      v = wemb[(size_t)sid * EMB + d];
      if (valid) vr = wemb[(size_t)sid_r * EMB + d];
    } else if (d < DTOT) {
      v = memb[mid * MDIM + (d - EMB)];
      if (valid) vr = memb[mid_r * MDIM + (d - EMB)];
    }
    xbf[base + d]  = f2bf(v);
    xrbf[base + d] = f2bf(vr);
  }
}

// Transpose weight [G, K] f32 -> K-major bf16 [Kp, G] (WMMA B-matrix friendly)
__global__ void wtrans_kernel(const float* __restrict__ w,
                              unsigned short* __restrict__ wT,
                              int K, int Kp) {
  size_t n_total = (size_t)Kp * Gn;
  size_t i = (size_t)blockIdx.x * blockDim.x + threadIdx.x;
  size_t s = (size_t)gridDim.x * blockDim.x;
  for (; i < n_total; i += s) {
    int k = (int)(i / Gn);
    int n = (int)(i % Gn);
    wT[i] = (k < K) ? f2bf(w[(size_t)n * K + k]) : (unsigned short)0;
  }
}

// ---------------- hoisted input projection: xg = x @ Wih^T ----------------
// Grid: (G/128, B/16, T); 256 threads = 8 waves; one 16x16 tile per wave.
// The shared 16x352 A tile is staged once per block by the TDM; output layout
// [T, G, B] bf16 so each lane's C-fragment is a single 16-byte store.
__global__ __launch_bounds__(256) void xg_gemm_kernel(
    const unsigned short* __restrict__ xbf,  // [B*T, KP]
    const unsigned short* __restrict__ wT,   // [KP, G]
    unsigned short* __restrict__ xg)         // [T, G, B]
{
#if HAVE_TDM
  __shared__ __align__(16) unsigned short a_lds[16 * KP];  // 11 KB
#endif
  const int t    = blockIdx.z;
  const int b0   = blockIdx.y * 16;
  const int wv   = threadIdx.x >> 5;
  const int n0   = blockIdx.x * 128 + wv * 16;
  const int lane = threadIdx.x & 31;
  const int half = lane >> 4, l15 = lane & 15;

#if HAVE_TDM
  if (wv == 0) {
    // rows b0..b0+15 of x at step t; line stride = Tn*KP elements
    tdm_load_2d(xbf, (((size_t)b0 * Tn + t) * KP) * 2ull, &a_lds[0],
                /*tile*/ KP, 16, /*tensor*/ KP, (unsigned)(Bn * Tn),
                /*stride0*/ (unsigned)(Tn * KP));
    __builtin_amdgcn_s_wait_tensorcnt(0);
  }
  __syncthreads();
#endif

  v8f acc = {};
#if !HAVE_TDM
  const size_t rowbase = ((size_t)(b0 + l15) * Tn + t) * KP;
#endif
#pragma unroll 1
  for (int klo = 0; klo < KP; klo += 32) {
    FragBF a, b;
#if HAVE_TDM
    const uint4* ap =
        reinterpret_cast<const uint4*>(a_lds + (size_t)l15 * KP + klo + half * 8);
#else
    const uint4* ap = reinterpret_cast<const uint4*>(xbf + rowbase + klo + half * 8);
#endif
    a.u[0] = ap[0];
    a.u[1] = ap[2];
    {
      unsigned short bt[16];
#pragma unroll
      for (int s = 0; s < 16; ++s) {
        int kb = klo + ((s < 8) ? (half * 8 + s) : (16 + half * 8 + s - 8));
        bt[s] = wT[(size_t)kb * Gn + n0 + l15];
      }
      __builtin_memcpy(&b, bt, 32);
    }
    if (klo + 32 < KP)
      __builtin_prefetch(wT + (size_t)(klo + 32 + lane) * Gn + n0, 0, 0);
    acc = wmma_bf16(a.v, b.v, acc);
  }
  unsigned short pk[8];
#pragma unroll
  for (int r = 0; r < 8; ++r) pk[r] = f2bf(acc[r]);
  uint4 q;
  __builtin_memcpy(&q, pk, 16);
  *reinterpret_cast<uint4*>(xg + ((size_t)t * Gn + n0 + l15) * Bn + b0 + half * 8) = q;
}

// ---------------- persistent GRU recurrence (one direction) ----------------
// 16 blocks x 1024 threads (32 waves); block owns 16 batch rows; wave w owns
// gate columns j in [16w, 16w+16) across r/z/n so gate math is wave-local.
// f32 master h lives in per-lane registers (owning lane is step-invariant);
// bf16 h in LDS for WMMA A-fragments. The per-step xg tile is double-buffered
// via the Tensor Data Mover so the DMA overlaps the 48-WMMA compute phase.
__global__ __launch_bounds__(1024) void gru_kernel(
    const unsigned short* __restrict__ xg,    // [T, G, B] bf16 (pre-projected)
    const unsigned short* __restrict__ whhT,  // [HD, G] bf16 K-major
    const float* __restrict__ bih,
    const float* __restrict__ bhh,
    const int* __restrict__ lens,
    float* __restrict__ ctx,                  // [B, T, H] f32 (tanh fused)
    int coloff, int backward)
{
  __shared__ __align__(16) unsigned short h_bf[16 * HDn];     // 16 KB
#if HAVE_TDM
  __shared__ __align__(16) unsigned short xg_lds[2][Gn * 16]; // 2x48 KB
#endif
  __shared__ int lens_s[16];

  const int tid  = threadIdx.x;
  const int wv   = tid >> 5;        // 0..31
  const int lane = tid & 31;
  const int half = lane >> 4;
  const int l15  = lane & 15;
  const int b0   = blockIdx.x * 16;
  const int j    = 16 * wv + l15;   // this lane's hidden-dim column

  for (int i = tid; i < 16 * HDn; i += blockDim.x) h_bf[i] = 0;
  if (tid < 16) lens_s[tid] = lens[b0 + tid];

  const float bihr = bih[j],        bhhr = bhh[j];
  const float bihz = bih[512 + j],  bhhz = bhh[512 + j];
  const float bihn = bih[1024 + j], bhhn = bhh[1024 + j];
  const int n0 = 16 * wv;
  float hreg[8] = {0.f, 0.f, 0.f, 0.f, 0.f, 0.f, 0.f, 0.f};  // f32 master h

#if HAVE_TDM
  if (wv == 0) {
    tdm_load_2d(xg, (((size_t)0 * Gn) * Bn + b0) * 2ull, &xg_lds[0][0],
                /*tile*/ 16, Gn, /*tensor*/ Bn, (unsigned)(Gn * Tn),
                /*stride0*/ Bn);
    __builtin_amdgcn_s_wait_tensorcnt(0);
  }
#endif
  __syncthreads();

  for (int t = 0; t < Tn; ++t) {
#if HAVE_TDM
    if (wv == 0 && (t + 1) < Tn) {
      // prefetch next step's gate tile into the other buffer (overlaps compute)
      tdm_load_2d(xg, (((size_t)(t + 1) * Gn) * Bn + b0) * 2ull,
                  &xg_lds[(t + 1) & 1][0],
                  16, Gn, Bn, (unsigned)(Gn * Tn), Bn);
    }
#endif

    // ---- hidden projection: accH = h @ Whh^T for columns [n0, n0+16) ----
    v8f accH[3] = {};
#pragma unroll 1
    for (int klo = 0; klo < HDn; klo += 32) {
      FragBF a;
      const uint4* ap =
          reinterpret_cast<const uint4*>(h_bf + (size_t)l15 * HDn + klo + half * 8);
      a.u[0] = ap[0];
      a.u[1] = ap[2];
      int kk = klo + lane;
      if (klo + 32 < HDn)
        __builtin_prefetch(whhT + (size_t)(kk + 32) * Gn + n0, 0, 0);
      FragBF br, bz, bn;
      {
        unsigned short tr_[16], tz_[16], tn_[16];
#pragma unroll
        for (int s = 0; s < 16; ++s) {
          int kb = klo + ((s < 8) ? (half * 8 + s) : (16 + half * 8 + s - 8));
          tr_[s] = whhT[(size_t)kb * Gn + n0 + l15];
          tz_[s] = whhT[(size_t)kb * Gn + 512 + n0 + l15];
          tn_[s] = whhT[(size_t)kb * Gn + 1024 + n0 + l15];
        }
        __builtin_memcpy(&br, tr_, 32);
        __builtin_memcpy(&bz, tz_, 32);
        __builtin_memcpy(&bn, tn_, 32);
      }
      accH[0] = wmma_bf16(a.v, br.v, accH[0]);
      accH[1] = wmma_bf16(a.v, bz.v, accH[1]);
      accH[2] = wmma_bf16(a.v, bn.v, accH[2]);
    }

    // ---- fetch pre-projected input gates for this lane's (m-slots, j) ----
    unsigned short xv[3][8];
#pragma unroll
    for (int r = 0; r < 3; ++r) {
#if HAVE_TDM
      uint4 q = *reinterpret_cast<const uint4*>(
          &xg_lds[t & 1][0] + ((size_t)(r * 512 + j)) * 16 + half * 8);
#else
      uint4 q = *reinterpret_cast<const uint4*>(
          xg + ((size_t)t * Gn + r * 512 + j) * Bn + b0 + half * 8);
#endif
      __builtin_memcpy(xv[r], &q, 16);
    }

    __syncthreads();  // all reads of h_bf / xg_lds[t&1] complete before update

    // ---- GRU gates + h update: lane owns column j, 8 m-slots ----
#pragma unroll
    for (int r8 = 0; r8 < 8; ++r8) {
      int m = half * 8 + r8;
      bool valid = (t < lens_s[m]);
      float xr = bf2f(xv[0][r8]) + bihr;
      float hr = accH[0][r8] + bhhr;
      float xz = bf2f(xv[1][r8]) + bihz;
      float hz = accH[1][r8] + bhhz;
      float xn = bf2f(xv[2][r8]) + bihn;
      float hn = accH[2][r8] + bhhn;
      float rg = 1.f / (1.f + expf(-(xr + hr)));
      float zg = 1.f / (1.f + expf(-(xz + hz)));
      float ng = tanhf(xn + rg * hn);
      float hold = hreg[r8];
      float hnew = valid ? ((1.f - zg) * ng + zg * hold) : hold;
      float outv = valid ? hnew : 0.f;
      hreg[r8] = hnew;
      h_bf[m * HDn + j] = f2bf(hnew);
      if (!backward) {
        ctx[((size_t)(b0 + m) * Tn + t) * Hn + coloff + j] = tanhf(outv);
      } else if (valid) {
        int to = lens_s[m] - 1 - t;  // rev_valid scatter back
        ctx[((size_t)(b0 + m) * Tn + to) * Hn + coloff + j] = tanhf(outv);
      }
    }

#if HAVE_TDM
    if (wv == 0) __builtin_amdgcn_s_wait_tensorcnt(0);  // next xg tile resident
#endif
    __syncthreads();  // h update + next xg buffer visible before next step
  }
}

// ---------------- target-average pooling + broadcast add ----------------
__global__ void tavg_kernel(const int* __restrict__ masks,
                            const float* __restrict__ ctx,
                            float* __restrict__ tavg) {
  int b = blockIdx.x;
  for (int d = threadIdx.x; d < Hn; d += blockDim.x) {
    float s = 0.f, cnt = 0.f;
    for (int t = 0; t < Tn; ++t) {
      float mf = (float)masks[b * Tn + t];
      s += mf * ctx[((size_t)b * Tn + t) * Hn + d];
      cnt += mf;
    }
    tavg[(size_t)b * Hn + d] = s / cnt;
  }
}

__global__ void add_tavg_kernel(float* __restrict__ ctx, const float* __restrict__ tavg) {
  size_t n = (size_t)Bn * Tn * Hn;
  size_t i = (size_t)blockIdx.x * blockDim.x + threadIdx.x;
  size_t s = (size_t)gridDim.x * blockDim.x;
  for (; i < n; i += s) {
    size_t d = i % Hn;
    size_t b = i / ((size_t)Tn * Hn);
    ctx[i] += tavg[b * Hn + d];
  }
}

// ---------------- CRF feature projection ----------------
__global__ void feats_kernel(const float* __restrict__ ctx,
                             const float* __restrict__ tri_w,
                             const float* __restrict__ tri_b,
                             float* __restrict__ feats) {
  int bt = blockIdx.x;
  __shared__ float red[128][4];
  float acc[4] = {0.f, 0.f, 0.f, 0.f};
  for (int d = threadIdx.x; d < Hn; d += 128) {
    float c = ctx[(size_t)bt * Hn + d];
#pragma unroll
    for (int l = 0; l < 4; ++l) acc[l] += c * tri_w[d * 4 + l];
  }
#pragma unroll
  for (int l = 0; l < 4; ++l) red[threadIdx.x][l] = acc[l];
  __syncthreads();
  for (int s = 64; s > 0; s >>= 1) {
    if (threadIdx.x < s)
#pragma unroll
      for (int l = 0; l < 4; ++l) red[threadIdx.x][l] += red[threadIdx.x + s][l];
    __syncthreads();
  }
  if (threadIdx.x < 4)
    feats[(size_t)bt * 4 + threadIdx.x] = red[0][threadIdx.x] + tri_b[threadIdx.x];
}

// ---------------- CRF forward-backward (one thread per batch row) ----------------
__device__ inline float lse4(float a, float b, float c, float d) {
  float m = fmaxf(fmaxf(a, b), fmaxf(c, d));
  return m + logf(expf(a - m) + expf(b - m) + expf(c - m) + expf(d - m));
}

__global__ void crf_kernel(const float* __restrict__ feats,
                           const int* __restrict__ lens,
                           const float* __restrict__ trans,
                           float* __restrict__ alphas,  // [B,T,4]
                           float* __restrict__ sp) {    // [B,T]
  int b = blockIdx.x * blockDim.x + threadIdx.x;
  if (b >= Bn) return;
  float tr[16];
#pragma unroll
  for (int i = 0; i < 16; ++i) tr[i] = trans[i];
  int len = lens[b];

  float alpha[4];
#pragma unroll
  for (int l = 0; l < 4; ++l) alpha[l] = feats[((size_t)b * Tn + 0) * 4 + l];
#pragma unroll
  for (int l = 0; l < 4; ++l) alphas[((size_t)b * Tn + 0) * 4 + l] = alpha[l];
  for (int t = 1; t < Tn; ++t) {
    if (t < len) {
      float f[4], an[4];
#pragma unroll
      for (int l = 0; l < 4; ++l) f[l] = feats[((size_t)b * Tn + t) * 4 + l];
#pragma unroll
      for (int i = 0; i < 4; ++i)
        an[i] = f[i] + lse4(alpha[0] + tr[0 * 4 + i], alpha[1] + tr[1 * 4 + i],
                            alpha[2] + tr[2 * 4 + i], alpha[3] + tr[3 * 4 + i]);
#pragma unroll
      for (int l = 0; l < 4; ++l) alpha[l] = an[l];
    }
#pragma unroll
    for (int l = 0; l < 4; ++l) alphas[((size_t)b * Tn + t) * 4 + l] = alpha[l];
  }

  float beta[4] = {0.f, 0.f, 0.f, 0.f};
  for (int t = Tn - 1; t >= 0; --t) {
    if (t < Tn - 1 && (t + 1) < len) {
      float f1[4], bn[4];
#pragma unroll
      for (int l = 0; l < 4; ++l) f1[l] = feats[((size_t)b * Tn + t + 1) * 4 + l];
#pragma unroll
      for (int i = 0; i < 4; ++i)
        bn[i] = lse4(tr[i * 4 + 0] + f1[0] + beta[0], tr[i * 4 + 1] + f1[1] + beta[1],
                     tr[i * 4 + 2] + f1[2] + beta[2], tr[i * 4 + 3] + f1[3] + beta[3]);
#pragma unroll
      for (int l = 0; l < 4; ++l) beta[l] = bn[l];
    }
    float s0 = alphas[((size_t)b * Tn + t) * 4 + 0] + beta[0];
    float s1 = alphas[((size_t)b * Tn + t) * 4 + 1] + beta[1];
    float s2 = alphas[((size_t)b * Tn + t) * 4 + 2] + beta[2];
    float s3 = alphas[((size_t)b * Tn + t) * 4 + 3] + beta[3];
    float mx = fmaxf(fmaxf(s0, s1), fmaxf(s2, s3));
    float den = expf(s0 - mx) + expf(s1 - mx) + expf(s2 - mx) + expf(s3 - mx);
    float p = expf(s1 - mx) / den;
    sp[(size_t)b * Tn + t] = (t < len) ? p : 0.f;
  }
}

// ---------------- sentiment pooling + classifier (per batch row) ----------------
__global__ void sent_kernel(const float* __restrict__ sp,
                            const float* __restrict__ ctx,
                            const float* __restrict__ sent_w,
                            const float* __restrict__ sent_b,
                            const int* __restrict__ labels,
                            float* __restrict__ out_scores,  // d_out+2, [B,3]
                            float* __restrict__ pb_cls,
                            float* __restrict__ pb_nrm) {
  int b = blockIdx.x;
  int tid = threadIdx.x;
  __shared__ float red[256];
  __shared__ float svs[Hn];
  __shared__ float ls[3];

  float s = (tid < Tn) ? sp[(size_t)b * Tn + tid] : 0.f;
  red[tid] = s;
  __syncthreads();
  for (int st = 128; st > 0; st >>= 1) {
    if (tid < st) red[tid] += red[tid + st];
    __syncthreads();
  }
  float spsum = red[0];
  float gamma = spsum * 0.5f;

  for (int d = tid; d < Hn; d += 256) {
    float acc = 0.f;
    for (int t = 0; t < Tn; ++t)
      acc += sp[(size_t)b * Tn + t] * ctx[((size_t)b * Tn + t) * Hn + d];
    svs[d] = acc / gamma;
  }
  __syncthreads();
  if (tid < 3) {
    float acc = 0.f;
    for (int d = 0; d < Hn; ++d) acc += svs[d] * sent_w[d * 3 + tid];
    ls[tid] = acc + sent_b[tid];
  }
  __syncthreads();
  if (tid == 0) {
    float mx = fmaxf(ls[0], fmaxf(ls[1], ls[2]));
    float lse = mx + logf(expf(ls[0] - mx) + expf(ls[1] - mx) + expf(ls[2] - mx));
    float sc0 = ls[0] - lse, sc1 = ls[1] - lse, sc2 = ls[2] - lse;
    out_scores[b * 3 + 0] = sc0;
    out_scores[b * 3 + 1] = sc1;
    out_scores[b * 3 + 2] = sc2;
    int lb = labels[b];
    float chosen = (lb == 0) ? sc0 : ((lb == 1) ? sc1 : sc2);
    pb_cls[b] = -chosen;
    pb_nrm[b] = spsum;
  }
}

// deterministic final reduction (no float atomics)
__global__ void final_kernel(const float* __restrict__ pb_cls,
                             const float* __restrict__ pb_nrm,
                             const float* __restrict__ trans,
                             float* __restrict__ out) {
  if (blockIdx.x == 0 && threadIdx.x == 0) {
    float c = 0.f, n = 0.f;
    for (int b = 0; b < Bn; ++b) { c += pb_cls[b]; n += pb_nrm[b]; }
    float pena = fmaxf(trans[1 * 4 + 0] - trans[0 * 4 + 0], 0.f) +
                 fmaxf(trans[0 * 4 + 1] - trans[1 * 4 + 1], 0.f);
    out[0] = c / (float)Bn;
    out[1] = C1 * pena + C2 * (n / (float)Bn);
  }
}

// ---------------- host launcher ----------------
extern "C" void kernel_launch(void* const* d_in, const int* in_sizes, int n_in,
                              void* d_out, int out_size, void* d_ws, size_t ws_size,
                              hipStream_t stream) {
  const int*   sents  = (const int*)d_in[0];
  const int*   masks  = (const int*)d_in[1];
  const int*   labels = (const int*)d_in[2];
  const int*   lens   = (const int*)d_in[3];
  const float* wemb   = (const float*)d_in[4];
  const float* memb   = (const float*)d_in[5];
  const float* w_ih_f = (const float*)d_in[6];
  const float* w_hh_f = (const float*)d_in[7];
  const float* b_ih_f = (const float*)d_in[8];
  const float* b_hh_f = (const float*)d_in[9];
  const float* w_ih_b = (const float*)d_in[10];
  const float* w_hh_b = (const float*)d_in[11];
  const float* b_ih_b = (const float*)d_in[12];
  const float* b_hh_b = (const float*)d_in[13];
  const float* tri_w  = (const float*)d_in[14];
  const float* tri_b  = (const float*)d_in[15];
  const float* trans  = (const float*)d_in[16];
  const float* sent_w = (const float*)d_in[17];
  const float* sent_b = (const float*)d_in[18];
  float* out = (float*)d_out;

  // workspace carve-out
  char* ws = (char*)d_ws;
  size_t off = 0;
  auto carve = [&](size_t bytes) -> char* {
    off = (off + 255) & ~(size_t)255;
    char* p = ws + off;
    off += bytes;
    return p;
  };
  unsigned short* xbf   = (unsigned short*)carve((size_t)Bn * Tn * KP * 2);
  unsigned short* xrbf  = (unsigned short*)carve((size_t)Bn * Tn * KP * 2);
  unsigned short* wihfT = (unsigned short*)carve((size_t)KP * Gn * 2);
  unsigned short* wihbT = (unsigned short*)carve((size_t)KP * Gn * 2);
  unsigned short* whhfT = (unsigned short*)carve((size_t)HDn * Gn * 2);
  unsigned short* whhbT = (unsigned short*)carve((size_t)HDn * Gn * 2);
  unsigned short* xgf   = (unsigned short*)carve((size_t)Tn * Gn * Bn * 2);
  unsigned short* xgb   = (unsigned short*)carve((size_t)Tn * Gn * Bn * 2);
  float* ctx    = (float*)carve((size_t)Bn * Tn * Hn * 4);
  float* tavg   = (float*)carve((size_t)Bn * Hn * 4);
  float* feats  = (float*)carve((size_t)Bn * Tn * 4 * 4);
  float* alphas = (float*)carve((size_t)Bn * Tn * 4 * 4);
  float* spbuf  = (float*)carve((size_t)Bn * Tn * 4);
  float* pb_cls = (float*)carve((size_t)Bn * 4);
  float* pb_nrm = (float*)carve((size_t)Bn * 4);

  // 1) zero ctx (backward half at padded t is never written otherwise)
  zero_f32_kernel<<<2048, 256, 0, stream>>>(ctx, (size_t)Bn * Tn * Hn);

  // 2) embeddings (forward + reversed-valid), bf16
  embed_kernel<<<Bn * Tn, 128, 0, stream>>>(sents, masks, lens, wemb, memb, xbf, xrbf);

  // 3) weights -> K-major bf16
  wtrans_kernel<<<2112, 256, 0, stream>>>(w_ih_f, wihfT, DTOT, KP);
  wtrans_kernel<<<2112, 256, 0, stream>>>(w_ih_b, wihbT, DTOT, KP);
  wtrans_kernel<<<3072, 256, 0, stream>>>(w_hh_f, whhfT, HDn, HDn);
  wtrans_kernel<<<3072, 256, 0, stream>>>(w_hh_b, whhbT, HDn, HDn);

  // 4) hoisted input projections (massively parallel WMMA GEMMs, TDM-staged A)
  dim3 ggrid(Gn / 128, Bn / 16, Tn);
  xg_gemm_kernel<<<ggrid, 256, 0, stream>>>(xbf,  wihfT, xgf);
  xg_gemm_kernel<<<ggrid, 256, 0, stream>>>(xrbf, wihbT, xgb);

  // 5) bi-GRU recurrence, persistent blocks over batch tiles
  gru_kernel<<<16, 1024, 0, stream>>>(xgf, whhfT, b_ih_f, b_hh_f, lens, ctx, 0,   0);
  gru_kernel<<<16, 1024, 0, stream>>>(xgb, whhbT, b_ih_b, b_hh_b, lens, ctx, HDn, 1);

  // 6) target average + broadcast add
  tavg_kernel<<<Bn, 256, 0, stream>>>(masks, ctx, tavg);
  add_tavg_kernel<<<2048, 256, 0, stream>>>(ctx, tavg);

  // 7) CRF features
  feats_kernel<<<Bn * Tn, 128, 0, stream>>>(ctx, tri_w, tri_b, feats);

  // 8) CRF forward-backward marginals (state 1 prob)
  crf_kernel<<<1, 256, 0, stream>>>(feats, lens, trans, alphas, spbuf);

  // 9) sentiment pooling + classifier + per-row partials
  sent_kernel<<<Bn, 256, 0, stream>>>(spbuf, ctx, sent_w, sent_b, labels,
                                      out + 2, pb_cls, pb_nrm);

  // 10) deterministic scalar reduction -> cls_loss, norm_pen
  final_kernel<<<1, 1, 0, stream>>>(pb_cls, pb_nrm, trans, out);
}
